// SPTModule_24927990186527
// MI455X (gfx1250) — matrix operand order repaired and checked by reading, not compiled, not run on hardware
//
#include <hip/hip_runtime.h>
#include <hip/hip_bf16.h>
#include <math.h>

// ---------------- CDNA5 WMMA types ----------------
typedef __attribute__((ext_vector_type(16))) __bf16 v16bf;
typedef __attribute__((ext_vector_type(8)))  float  v8f;
typedef __attribute__((__vector_size__(16))) int    v4i;   // matches builtin param type

union Frag16 { uint4 u[2]; v16bf v; };

#define NEG_BIG  (-3.0e38f)
#define DEAD_THR (-1.0e37f)

static __device__ __forceinline__ v8f wmma_bf16(v16bf a, v16bf b, v8f c) {
  return __builtin_amdgcn_wmma_f32_16x16x32_bf16(false, a, false, b, (short)0, c, false, false);
}

// Async global->LDS path (gfx1250 ASYNCcnt engine), guarded so compilation
// succeeds on toolchains without the builtin (falls back to plain LDS copy).
#if defined(__HIP_DEVICE_COMPILE__) && defined(__gfx1250__) && \
    __has_builtin(__builtin_amdgcn_global_load_async_to_lds_b128)
#define USE_ASYNC_LDS 1
typedef __attribute__((address_space(1))) v4i* gas_b128;
typedef __attribute__((address_space(3))) v4i* las_b128;
#else
#define USE_ASYNC_LDS 0
#endif

static __device__ __forceinline__ void wait_async_all() {
#if defined(__HIP_DEVICE_COMPILE__) && defined(__gfx1250__)
#if __has_builtin(__builtin_amdgcn_s_wait_asynccnt)
  __builtin_amdgcn_s_wait_asynccnt(0);
#else
  asm volatile("s_wait_asynccnt 0x0" ::: "memory");
#endif
#endif
}

// ---------------- bilinear resize [B,C,H,W] f32 -> tokens [B,h*w,C] bf16 ----------------
__global__ __launch_bounds__(256)
void k_resize_tokens(const float* __restrict__ x, __bf16* __restrict__ out,
                     int B, int C, int H, int W, int h, int w) {
  int total = B * h * w * C;
  int i = blockIdx.x * blockDim.x + threadIdx.x;
  if (i >= total) return;
  int c = i % C; int t = i / C;
  int xo = t % w; t /= w;
  int yo = t % h; int b = t / h;
  float ry = (h > 1) ? (float)(H - 1) / (float)(h - 1) : 0.f;
  float rx = (w > 1) ? (float)(W - 1) / (float)(w - 1) : 0.f;
  float fy = yo * ry; int y0 = (int)fy; float wy = fy - y0; int y1 = min(y0 + 1, H - 1);
  float fx = xo * rx; int x0 = (int)fx; float wx = fx - x0; int x1 = min(x0 + 1, W - 1);
  const float* base = x + ((size_t)b * C + c) * H * W;
  float v00 = base[y0 * W + x0], v01 = base[y0 * W + x1];
  float v10 = base[y1 * W + x0], v11 = base[y1 * W + x1];
  float v = (1.f - wy) * ((1.f - wx) * v00 + wx * v01) + wy * ((1.f - wx) * v10 + wx * v11);
  out[i] = (__bf16)v;
}

// ---------------- weight convert: f32 W -> bf16 Wt[N][K] ----------------
// src_is_KN=1: W is [K,N] (qw/kw/vw/ow, in x out) -> transpose.
// src_is_KN=0: W is [N,K] (scale_w/cf_w, out x in)  -> copy.
__global__ __launch_bounds__(256)
void k_weight_conv(const float* __restrict__ W, __bf16* __restrict__ Wt,
                   int N, int K, int src_is_KN) {
  int i = blockIdx.x * blockDim.x + threadIdx.x;
  if (i >= N * K) return;
  int n = i / K, kk = i % K;
  float v = src_is_KN ? W[(size_t)kk * N + n] : W[(size_t)n * K + kk];
  Wt[i] = (__bf16)v;
}

// ---------------- WMMA GEMM with LDS-staged weights ----------------
// out[m,n] = A[m,:]·Wt[n,:] + bias[n].  A bf16 [M,K] row-major, Wt bf16 [N][K]
// (weights pre-transposed). K%32==0, M%16==0, N%16==0.
// Block = 8 waves; whole Wt (N*K bf16, <=96KB) staged to LDS via async
// global->LDS; each wave owns one 16-row M-tile and sweeps all N-tiles, so A
// fragments come from near caches and B fragments from LDS.
__global__ __launch_bounds__(256)
void k_gemm_wmma(const __bf16* __restrict__ A, const __bf16* __restrict__ Wt,
                 const float* __restrict__ bias,
                 __bf16* __restrict__ outBf, float* __restrict__ outF,
                 int M, int N, int K) {
  extern __shared__ __bf16 Wlds[];

  // ---- stage weights into LDS (all threads participate before any exit) ----
  int totalVec = (N * K) >> 3;              // 16-byte units
  for (int idx = threadIdx.x; idx < totalVec; idx += blockDim.x) {
#if USE_ASYNC_LDS
    __builtin_amdgcn_global_load_async_to_lds_b128(
        (gas_b128)(Wt + (size_t)idx * 8),
        (las_b128)(Wlds + (size_t)idx * 8),
        0, 0);
#else
    *(uint4*)(Wlds + (size_t)idx * 8) = *(const uint4*)(Wt + (size_t)idx * 8);
#endif
  }
#if USE_ASYNC_LDS
  wait_async_all();
#endif
  __syncthreads();

  int wave = threadIdx.x >> 5;
  int lane = threadIdx.x & 31;
  int lan  = lane & 15;
  int mt = blockIdx.x * 8 + wave;
  if (mt >= (M >> 4)) return;               // wave-uniform, after the barrier

  int koffA = (lane < 16) ? 0 : 8;          // A frag: k {0..7,16..23}/{8..15,24..31}
  int koffB = (lane < 16) ? 0 : 8;          // B frag: k {0..7,16..23}/{8..15,24..31}
  int half8 = (lane < 16) ? 0 : 8;
  const __bf16* Arow = A + (size_t)(mt * 16 + lan) * K;
  __builtin_prefetch(Arow, 0, 1);           // -> global_prefetch_b8

  int tilesN = N >> 4;
  for (int nt = 0; nt < tilesN; ++nt) {
    const __bf16* Wrow = Wlds + (size_t)(nt * 16 + lan) * K;
    v8f acc = {0.f,0.f,0.f,0.f,0.f,0.f,0.f,0.f};
    for (int kb = 0; kb < K; kb += 32) {
      Frag16 af, bf;
      af.u[0] = *(const uint4*)(Arow + kb + koffA);
      af.u[1] = *(const uint4*)(Arow + kb + koffA + 16);
      bf.u[0] = *(const uint4*)(Wrow + kb + koffB);      // ds_load_b128
      bf.u[1] = *(const uint4*)(Wrow + kb + koffB + 16);
      acc = wmma_bf16(af.v, bf.v, acc);
    }
    float bv = bias ? bias[nt * 16 + lan] : 0.f;
    int ncol = nt * 16 + lan;
#pragma unroll
    for (int r = 0; r < 8; ++r) {
      int m = mt * 16 + half8 + r;
      float v = acc[r] + bv;
      if (outBf) outBf[(size_t)m * N + ncol] = (__bf16)v;
      if (outF)  outF [(size_t)m * N + ncol] = v;
    }
  }
}

// ---------------- meanQ[b,h,d] = mean_n Q[b,n,h*D+d] ----------------
__global__ __launch_bounds__(256)
void k_meanq(const __bf16* __restrict__ Qb, float* __restrict__ meanQ,
             int B, int NH, int Ntok, int C, int D) {
  int i = blockIdx.x * blockDim.x + threadIdx.x;
  if (i >= B * NH * D) return;
  int d = i % D; int h = (i / D) % NH; int b = i / (D * NH);
  const __bf16* p = Qb + (size_t)b * Ntok * C + h * D + d;
  float s = 0.f;
  for (int n = 0; n < Ntok; ++n) s += (float)p[(size_t)n * C];
  meanQ[i] = s / (float)Ntok;
}

// ---------------- colmean[b,h,m] = meanQ · K[m] / sqrt(D) ----------------
__global__ __launch_bounds__(256)
void k_colmean(const __bf16* __restrict__ Kb, const float* __restrict__ meanQ,
               float* __restrict__ colmean, int B, int NH, int Ntok, int C, int D,
               float invsqd) {
  int i = blockIdx.x * blockDim.x + threadIdx.x;
  if (i >= B * NH * Ntok) return;
  int m = i % Ntok; int h = (i / Ntok) % NH; int b = i / (Ntok * NH);
  const __bf16* kr = Kb + ((size_t)b * Ntok + m) * C + h * D;
  const float* mq = meanQ + (b * NH + h) * D;
  float s = 0.f;
#pragma unroll
  for (int d = 0; d < 32; ++d) s += mq[d] * (float)kr[d];
  colmean[i] = s * invsqd;
}

// ---------------- top-k via bitonic sort in LDS; one block per (b,h) ----------------
__global__ __launch_bounds__(256)
void k_topk(const float* __restrict__ colmean, int* __restrict__ sel,
            int Ntok, int P2, int kact, int kpad) {
  __shared__ float val[4096];
  __shared__ int   idx[4096];
  int bh = blockIdx.x;
  const float* cm = colmean + (size_t)bh * Ntok;
  for (int i = threadIdx.x; i < P2; i += blockDim.x) {
    val[i] = (i < Ntok) ? cm[i] : NEG_BIG;
    idx[i] = i;
  }
  __syncthreads();
  for (int ks = 2; ks <= P2; ks <<= 1) {
    for (int j = ks >> 1; j > 0; j >>= 1) {
      for (int i = threadIdx.x; i < P2; i += blockDim.x) {
        int ixj = i ^ j;
        if (ixj > i) {
          bool up = ((i & ks) == 0);           // descending
          float vi = val[i], vj = val[ixj];
          bool sw = up ? (vi < vj) : (vi > vj);
          if (sw) {
            val[i] = vj; val[ixj] = vi;
            int t = idx[i]; idx[i] = idx[ixj]; idx[ixj] = t;
          }
        }
      }
      __syncthreads();
    }
  }
  int* so = sel + (size_t)bh * kpad;
  for (int i = threadIdx.x; i < kpad; i += blockDim.x)
    so[i] = (i < kact) ? idx[i] : -1;
}

// ---------------- gather selected K (row-major [kpad][D]) and V transposed ([D][kpad]) ----------------
__global__ __launch_bounds__(256)
void k_gather(const __bf16* __restrict__ Kb, const __bf16* __restrict__ Vb,
              const int* __restrict__ sel, __bf16* __restrict__ Ksel,
              __bf16* __restrict__ Vtr, int B, int NH, int Ntok, int C, int D, int kpad) {
  int total = B * NH * kpad * D;
  int i = blockIdx.x * blockDim.x + threadIdx.x;
  if (i >= total) return;
  int d = i % D; int kk = (i / D) % kpad;
  int h = (i / (D * kpad)) % NH; int b = i / (D * kpad * NH);
  int s = sel[((size_t)b * NH + h) * kpad + kk];
  __bf16 kv = (__bf16)0.f, vv = (__bf16)0.f;
  if (s >= 0) {
    size_t off = ((size_t)b * Ntok + s) * C + h * D + d;
    kv = Kb[off];
    vv = Vb[off];
  }
  size_t bh = (size_t)b * NH + h;
  Ksel[(bh * kpad + kk) * D + d] = kv;
  Vtr [(bh * D + d) * kpad + kk] = vv;
}

// ---------------- flash attention over selected keys ----------------
// grid = (ceil(Ntok/16/8), B*NH), block = 256 (8 waves; each wave = one 16-query tile).
__global__ __launch_bounds__(256)
void k_flash(const __bf16* __restrict__ Qb, const __bf16* __restrict__ Ksel,
             const __bf16* __restrict__ Vtr, __bf16* __restrict__ Ob,
             int B, int NH, int Ntok, int C, int D, int kpad, int kact, float scale) {
  __shared__ __attribute__((aligned(16))) __bf16 Plds[8][16][32];
  int wave = threadIdx.x >> 5, lane = threadIdx.x & 31;
  int bh = blockIdx.y; int b = bh / NH, h = bh % NH;
  int nQT = Ntok >> 4;
  int qt = blockIdx.x * 8 + wave;
  if (qt >= nQT) return;                       // wave-uniform exit (EXEC stays all-ones)
  int lan = lane & 15;
  int half8 = (lane < 16) ? 0 : 8;
  int koffA = (lane < 16) ? 0 : 8;
  int koffB = (lane < 16) ? 0 : 8;
  int qbase = qt * 16;

  // Q A-fragment: row = qbase + lan, d chunks per A layout
  const __bf16* Qrow = Qb + ((size_t)b * Ntok + qbase + lan) * C + h * D;
  Frag16 aQ;
  aQ.u[0] = *(const uint4*)(Qrow + koffA);
  aQ.u[1] = *(const uint4*)(Qrow + koffA + 16);

  const __bf16* Kbase = Ksel + (size_t)bh * kpad * D;
  const __bf16* Vbase = Vtr  + (size_t)bh * D * kpad;

  v8f o0 = {0.f,0.f,0.f,0.f,0.f,0.f,0.f,0.f};
  v8f o1 = {0.f,0.f,0.f,0.f,0.f,0.f,0.f,0.f};
  float mrow[8], lrow[8];
#pragma unroll
  for (int r = 0; r < 8; ++r) { mrow[r] = NEG_BIG; lrow[r] = 0.f; }

  for (int kv = 0; kv < kpad; kv += 32) {
    // --- scores: two 16x16 tiles via 2 WMMAs ---
    Frag16 bK0, bK1;
    const __bf16* k0 = Kbase + (size_t)(kv + lan) * D + koffB;
    const __bf16* k1 = Kbase + (size_t)(kv + 16 + lan) * D + koffB;
    bK0.u[0] = *(const uint4*)(k0); bK0.u[1] = *(const uint4*)(k0 + 16);
    bK1.u[0] = *(const uint4*)(k1); bK1.u[1] = *(const uint4*)(k1 + 16);
    v8f zf = {0.f,0.f,0.f,0.f,0.f,0.f,0.f,0.f};
    v8f s0 = wmma_bf16(aQ.v, bK0.v, zf);
    v8f s1 = wmma_bf16(aQ.v, bK1.v, zf);
    int c0 = kv + lan, c1 = kv + 16 + lan;

    // --- online softmax per row (rows live in the same (vgpr,half) slots as C frags) ---
#pragma unroll
    for (int r = 0; r < 8; ++r) {
      float a = (c0 < kact) ? s0[r] * scale : NEG_BIG;
      float e = (c1 < kact) ? s1[r] * scale : NEG_BIG;
      float mx = fmaxf(a, e);
      mx = fmaxf(mx, __shfl_xor(mx, 1, 32));
      mx = fmaxf(mx, __shfl_xor(mx, 2, 32));
      mx = fmaxf(mx, __shfl_xor(mx, 4, 32));
      mx = fmaxf(mx, __shfl_xor(mx, 8, 32));
      float mnew = fmaxf(mrow[r], mx);
      bool dead = (mnew < DEAD_THR);
      float corr = dead ? 1.f : ((mrow[r] < DEAD_THR) ? 0.f : __expf(mrow[r] - mnew));
      float p0 = (dead || a < DEAD_THR) ? 0.f : __expf(a - mnew);
      float p1 = (dead || e < DEAD_THR) ? 0.f : __expf(e - mnew);
      float rs = p0 + p1;
      rs += __shfl_xor(rs, 1, 32);
      rs += __shfl_xor(rs, 2, 32);
      rs += __shfl_xor(rs, 4, 32);
      rs += __shfl_xor(rs, 8, 32);
      lrow[r] = lrow[r] * corr + rs;
      o0[r] *= corr; o1[r] *= corr;
      mrow[r] = mnew;
      Plds[wave][half8 + r][lan]      = (__bf16)p0;
      Plds[wave][half8 + r][16 + lan] = (__bf16)p1;
    }
    asm volatile("s_wait_dscnt 0x0" ::: "memory");   // wave-local LDS visibility

    // --- P (16x32) as A-fragment from LDS, PV via 2 WMMAs ---
    Frag16 aP;
    const __bf16* prow = &Plds[wave][lan][0];
    aP.u[0] = *(const uint4*)(prow + koffA);
    aP.u[1] = *(const uint4*)(prow + koffA + 16);
    Frag16 bV0, bV1;
    const __bf16* v0 = Vbase + (size_t)lan        * kpad + kv + koffB;
    const __bf16* v1 = Vbase + (size_t)(16 + lan) * kpad + kv + koffB;
    bV0.u[0] = *(const uint4*)(v0); bV0.u[1] = *(const uint4*)(v0 + 16);
    bV1.u[0] = *(const uint4*)(v1); bV1.u[1] = *(const uint4*)(v1 + 16);
    o0 = wmma_bf16(aP.v, bV0.v, o0);
    o1 = wmma_bf16(aP.v, bV1.v, o1);
  }

  // --- normalize, write O[b, q, h*D + d] ---
#pragma unroll
  for (int r = 0; r < 8; ++r) {
    float inv = (lrow[r] > 0.f) ? 1.f / lrow[r] : 0.f;
    int q = qbase + half8 + r;
    size_t base = ((size_t)b * Ntok + q) * C + h * D;
    Ob[base + lan]      = (__bf16)(o0[r] * inv);
    Ob[base + 16 + lan] = (__bf16)(o1[r] * inv);
  }
}

// ---------------- upsample tokens [B,h,w,C] -> fusedA[B*H*W][fusedK] at column colOff ----------------
__global__ __launch_bounds__(256)
void k_upsample(const __bf16* __restrict__ Fb, __bf16* __restrict__ fusedA,
                int B, int C, int h, int w, int H, int W, int colOff, int fusedK) {
  int total = B * H * W * C;
  int i = blockIdx.x * blockDim.x + threadIdx.x;
  if (i >= total) return;
  int c = i % C; int t = i / C;
  int X = t % W; t /= W;
  int Y = t % H; int b = t / H;
  float ry = (H > 1) ? (float)(h - 1) / (float)(H - 1) : 0.f;
  float rx = (W > 1) ? (float)(w - 1) / (float)(W - 1) : 0.f;
  float fy = Y * ry; int y0 = (int)fy; float wy = fy - y0; int y1 = min(y0 + 1, h - 1);
  float fx = X * rx; int x0 = (int)fx; float wx = fx - x0; int x1 = min(x0 + 1, w - 1);
  const __bf16* base = Fb + (size_t)b * h * w * C + c;
  float v00 = (float)base[((size_t)y0 * w + x0) * C];
  float v01 = (float)base[((size_t)y0 * w + x1) * C];
  float v10 = (float)base[((size_t)y1 * w + x0) * C];
  float v11 = (float)base[((size_t)y1 * w + x1) * C];
  float v = (1.f - wy) * ((1.f - wx) * v00 + wx * v01) + wy * ((1.f - wx) * v10 + wx * v11);
  fusedA[((size_t)b * H * W + Y * W + X) * fusedK + colOff + c] = (__bf16)v;
}

// ---------------- batch-norm stats: one block per channel ----------------
__global__ __launch_bounds__(256)
void k_bnstats(const float* __restrict__ fusedF, float* __restrict__ stats,
               int Mrows, int C) {
  __shared__ float s1[256], s2[256];
  int c = blockIdx.x;
  float a = 0.f, b = 0.f;
  for (int m = threadIdx.x; m < Mrows; m += blockDim.x) {
    float v = fusedF[(size_t)m * C + c];
    a += v; b += v * v;
  }
  s1[threadIdx.x] = a; s2[threadIdx.x] = b;
  __syncthreads();
  for (int s = 128; s > 0; s >>= 1) {
    if ((int)threadIdx.x < s) { s1[threadIdx.x] += s1[threadIdx.x + s]; s2[threadIdx.x] += s2[threadIdx.x + s]; }
    __syncthreads();
  }
  if (threadIdx.x == 0) {
    float mu = s1[0] / (float)Mrows;
    float var = s2[0] / (float)Mrows - mu * mu;
    stats[c] = mu;
    stats[C + c] = rsqrtf(var + 1e-5f);
  }
}

// ---------------- batch-norm apply, write [B,C,H,W] f32 ----------------
__global__ __launch_bounds__(256)
void k_bnapply(const float* __restrict__ fusedF, const float* __restrict__ stats,
               const float* __restrict__ gamma, const float* __restrict__ beta,
               float* __restrict__ out, int B, int C, int H, int W) {
  int total = B * C * H * W;
  int i = blockIdx.x * blockDim.x + threadIdx.x;
  if (i >= total) return;
  int xw = i % W; int t = i / W;
  int y = t % H; t /= H;
  int c = t % C; int b = t / C;
  float v = fusedF[((size_t)b * H * W + y * W + xw) * C + c];
  out[i] = gamma[c] * (v - stats[c]) * stats[C + c] + beta[c];
}

// =====================================================================
static inline int cdiv(int a, int b) { return (a + b - 1) / b; }

extern "C" void kernel_launch(void* const* d_in, const int* in_sizes, int n_in,
                              void* d_out, int out_size, void* d_ws, size_t ws_size,
                              hipStream_t stream) {
  (void)in_sizes; (void)n_in; (void)out_size; (void)ws_size;
  const int B = 2, C = 128, H = 48, W = 48, NH = 4, D = 32, L = 3;
  const int NTOK_MAX = H * W;                 // 2304
  const int M_MAX = B * NTOK_MAX;             // 4608
  const float invsqd = 0.17677669529663687f;  // 1/sqrt(32)

  const float* x       = (const float*)d_in[0];
  const float* scale_w = (const float*)d_in[1];
  const float* scale_b = (const float*)d_in[2];
  const float* qw = (const float*)d_in[3];  const float* qb = (const float*)d_in[4];
  const float* kw = (const float*)d_in[5];  const float* kb = (const float*)d_in[6];
  const float* vw = (const float*)d_in[7];  const float* vb = (const float*)d_in[8];
  const float* ow = (const float*)d_in[9];  const float* ob = (const float*)d_in[10];
  const float* cf_w = (const float*)d_in[11];
  const float* cf_b = (const float*)d_in[12];
  const float* bn_g = (const float*)d_in[13];
  const float* bn_b = (const float*)d_in[14];
  float* outp = (float*)d_out;

  // workspace carve-up (256B aligned slabs)
  char* wp = (char*)d_ws;
  auto carve = [&](size_t bytes) -> void* {
    void* p = (void*)wp;
    wp += (bytes + 255) & ~(size_t)255;
    return p;
  };
  __bf16* A_rs  = (__bf16*)carve((size_t)M_MAX * C * 2);
  __bf16* Fb    = (__bf16*)carve((size_t)M_MAX * C * 2);
  __bf16* Qb    = (__bf16*)carve((size_t)M_MAX * C * 2);
  __bf16* Kb    = (__bf16*)carve((size_t)M_MAX * C * 2);
  __bf16* Vb    = (__bf16*)carve((size_t)M_MAX * C * 2);
  __bf16* Obuf  = (__bf16*)carve((size_t)M_MAX * C * 2);
  __bf16* Wcvt  = (__bf16*)carve((size_t)3 * C * C * 2);      // up to [128][384]
  __bf16* Ksel  = (__bf16*)carve((size_t)B * NH * 1408 * D * 2);
  __bf16* Vtr   = (__bf16*)carve((size_t)B * NH * 1408 * D * 2);
  float*  meanQ = (float*) carve((size_t)B * NH * D * 4);
  float*  colm  = (float*) carve((size_t)B * NH * NTOK_MAX * 4);
  int*    sel   = (int*)   carve((size_t)B * NH * 1408 * 4);
  __bf16* fusA  = (__bf16*)carve((size_t)M_MAX * 3 * C * 2);
  float*  fusF  = (float*) carve((size_t)M_MAX * C * 4);
  float*  stats = (float*) carve((size_t)2 * C * 4);

  const int hs[3]  = {12, 24, 48};
  const int Ns[3]  = {144, 576, 2304};
  const int kas[3] = {86, 345, 1382};     // int(N*0.6)
  const int kps[3] = {96, 352, 1408};     // padded to 32
  const int P2s[3] = {256, 1024, 4096};

  auto gemm = [&](const __bf16* A, const float* Wsrc, int src_is_KN,
                  const float* bias, __bf16* oBf, float* oF, int M, int N, int K) {
    int wtot = N * K;
    k_weight_conv<<<cdiv(wtot, 256), 256, 0, stream>>>(Wsrc, Wcvt, N, K, src_is_KN);
    size_t ldsBytes = (size_t)N * K * 2;    // staged weights
    k_gemm_wmma<<<cdiv(M / 16, 8), 256, ldsBytes, stream>>>(A, Wcvt, bias, oBf, oF, M, N, K);
  };

  for (int si = 0; si < 3; ++si) {
    const int h = hs[si], w = hs[si];
    const int Ntok = Ns[si], M = B * Ntok;
    const int kact = kas[si], kpad = kps[si], P2 = P2s[si];

    // resize + scale projection -> Fb
    k_resize_tokens<<<cdiv(M * C, 256), 256, 0, stream>>>(x, A_rs, B, C, H, W, h, w);
    gemm(A_rs, scale_w + (size_t)si * C * C, /*src_is_KN=*/0,
         scale_b + (size_t)si * C, Fb, nullptr, M, C, C);

    for (int li = 0; li < L; ++li) {
      const size_t wo = (size_t)li * C * C;
      gemm(Fb, qw + wo, 1, qb + (size_t)li * C, Qb, nullptr, M, C, C);
      gemm(Fb, kw + wo, 1, kb + (size_t)li * C, Kb, nullptr, M, C, C);
      gemm(Fb, vw + wo, 1, vb + (size_t)li * C, Vb, nullptr, M, C, C);

      // top-k selection: mean-over-queries trick
      k_meanq<<<cdiv(B * NH * D, 256), 256, 0, stream>>>(Qb, meanQ, B, NH, Ntok, C, D);
      k_colmean<<<cdiv(B * NH * Ntok, 256), 256, 0, stream>>>(Kb, meanQ, colm,
                                                              B, NH, Ntok, C, D, invsqd);
      k_topk<<<B * NH, 256, 0, stream>>>(colm, sel, Ntok, P2, kact, kpad);
      k_gather<<<cdiv(B * NH * kpad * D, 256), 256, 0, stream>>>(
          Kb, Vb, sel, Ksel, Vtr, B, NH, Ntok, C, D, kpad);

      // flash attention over selected keys
      int nQT = Ntok / 16;
      dim3 fg(cdiv(nQT, 8), B * NH);
      k_flash<<<fg, 256, 0, stream>>>(Qb, Ksel, Vtr, Obuf,
                                      B, NH, Ntok, C, D, kpad, kact, invsqd);

      // output projection -> Fb (next layer input)
      gemm(Obuf, ow + wo, 1, ob + (size_t)li * C, Fb, nullptr, M, C, C);
    }

    // upsample into fused concat buffer at channel offset si*C
    k_upsample<<<cdiv(B * H * W * C, 256), 256, 0, stream>>>(
        Fb, fusA, B, C, h, w, H, W, si * C, 3 * C);
  }

  // fusion 1x1 conv (384 -> 128), f32 output for BN
  gemm(fusA, cf_w, /*src_is_KN=*/0, cf_b, nullptr, fusF, B * H * W, C, 3 * C);

  // batch norm
  k_bnstats<<<C, 256, 0, stream>>>(fusF, stats, B * H * W, C);
  k_bnapply<<<cdiv(B * C * H * W, 256), 256, 0, stream>>>(
      fusF, stats, bn_g, bn_b, outp, B, C, H, W);
}
